// MACE_7275674599653
// MI455X (gfx1250) — compile-verified, matrix operand order checked by reading, not run on hardware
//
#include <hip/hip_runtime.h>

typedef _Float16 v16h __attribute__((ext_vector_type(16)));
typedef _Float16 v8h  __attribute__((ext_vector_type(8)));
typedef float    v8f  __attribute__((ext_vector_type(8)));

#define DEVI __device__ __forceinline__

// ---------------- problem constants ----------------
constexpr int N_ATOMS = 4096;
constexpr int N_EDGES = 131072;
constexpr int HID     = 128;

// ---------------- workspace layout (bytes) ----------------
// weight panels in f16 "B-panel" format:
//   panel(kt,nt) = 512 halfs; lane l owns halfs [l*16 .. l*16+15] = its v16h
constexpr size_t OFF_PAN_RADW1   = 0;        // K=32 (src 8 padded), N=64
constexpr size_t OFF_PAN_RADW2   = 4096;     // K=64,  N=128
constexpr size_t OFF_PAN_TP      = 20480;    // 9 x (K=128,N=128)
constexpr size_t OFF_PAN_MSGW1   = 315392;   // K=192, N=128
constexpr size_t OFF_PAN_MSGW2   = 364544;   // K=128, N=128
constexpr size_t OFF_PAN_ATTNIN  = 397312;   // K=128, N=384 (transposed src)
constexpr size_t OFF_PAN_ATTNOUT = 495616;   // K=128, N=128 (transposed src)
constexpr size_t OFF_PAN_GATE    = 528384;
constexpr size_t OFF_PAN_OUT     = 561152;
constexpr size_t OFF_AGG         = 593920;   // f32 4096x128
constexpr size_t OFF_UPD         = 2691072;  // f32 4096x128
constexpr size_t OFF_UPD16       = 4788224;  // f16 4096x128
constexpr size_t OFF_QKV16       = 5836800;  // f16 4096x384
constexpr size_t OFF_ATT16       = 8982528;  // f16 4096x128

// ---------------- helpers ----------------
DEVI v8f zero8() {
  v8f z;
#pragma unroll
  for (int i = 0; i < 8; ++i) z[i] = 0.f;
  return z;
}

DEVI v8f wmma16(v16h a, v16h b, v8f c) {
  return __builtin_amdgcn_wmma_f32_16x16x32_f16(false, a, false, b, (short)0, c,
                                                false, false);
}

// A-operand (16x32 f16) gather from a row-major LDS tile.
// lane l -> row m=l&15; kh=l>>4; halfs 0..7 -> K=8*kh+h, halfs 8..15 ->
// K=16+8*kh+(h-8).  Both runs contiguous -> two ds_load_b128.
DEVI v16h lds_loadA(const _Float16* base, int ld, int lane, int koff) {
  int m = lane & 15, kh = (lane >> 4) & 1;
  const _Float16* p = base + m * ld + koff + kh * 8;
  v8h lo = *(const v8h*)(p);
  v8h hi = *(const v8h*)(p + 16);
  v16h a;
#pragma unroll
  for (int i = 0; i < 8; ++i) { a[i] = lo[i]; a[i + 8] = hi[i]; }
  return a;
}

DEVI void wait_ds() { asm volatile("s_wait_dscnt 0" ::: "memory"); }

DEVI float frcp(float x) { return __builtin_amdgcn_rcpf(x); }
DEVI float siluf(float x) { return x * frcp(1.f + __expf(-x)); }
DEVI float sigmf(float x) { return frcp(1.f + __expf(-x)); }

// ---------------- weight packing ----------------
__global__ void k_pack(_Float16* __restrict__ dst, const float* __restrict__ src,
                       int Ktot, int Ksrc, int N, int mode, int ldT, int s) {
  int nNt = N >> 4;
  int total = (Ktot >> 5) * nNt * 512;
  for (int idx = blockIdx.x * blockDim.x + threadIdx.x; idx < total;
       idx += gridDim.x * blockDim.x) {
    int p = idx >> 9;
    int lane = (idx >> 4) & 31;
    int h = idx & 15;
    int kt = p / nNt, nt = p - kt * nNt;
    int kh = lane >> 4;
    int kk = (h < 8) ? (8 * kh + h) : (16 + 8 * kh + (h - 8));
    int k = kt * 32 + kk;
    int n = nt * 16 + (lane & 15);
    float v = 0.f;
    if (mode == 0) {
      if (k < Ksrc) v = src[k * N + n];
    } else if (mode == 1) {
      v = src[n * ldT + k];
    } else {
      int row;
      if (s == 0)      row = k;
      else if (s < 4)  row = 128 + k * 3 + (s - 1);
      else             row = 512 + k * 5 + (s - 4);
      v = src[row * 128 + n];
    }
    dst[idx] = (_Float16)v;
  }
}

__global__ void k_zero(float* __restrict__ p, int n) {
  int i = blockIdx.x * blockDim.x + threadIdx.x;
  if (i < n) p[i] = 0.f;
}

// ------------- edge pipeline: 32 edges / wave (2 M-tiles share B) ---------
__global__ __launch_bounds__(128) void k_edge(
    const float* __restrict__ evec, const float* __restrict__ elen,
    const int* __restrict__ eidx, const float* __restrict__ rad_b1,
    const float* __restrict__ rad_b2, const float* __restrict__ tp_b,
    const _Float16* __restrict__ panW1, const _Float16* __restrict__ panW2,
    const _Float16* __restrict__ panTP, float* __restrict__ agg) {
  __shared__ _Float16 s_abuf[4][32 * 32];
  __shared__ _Float16 s_hbuf[4][32 * 64];
  __shared__ _Float16 s_rad[4][32 * 128];
  __shared__ float    s_sh[4][32 * 10];
  __shared__ int      s_dst[4][32];

  int wave = threadIdx.x >> 5, lane = threadIdx.x & 31;
  int nL = lane & 15, kh = (lane >> 4) & 1;
  int e0 = (blockIdx.x * 4 + wave) * 32;
  _Float16* abuf = s_abuf[wave];
  _Float16* hbuf = s_hbuf[wave];
  _Float16* radb = s_rad[wave];

  // ---- per-edge scalars: all 32 lanes, one edge each ----
  {
    int e = e0 + lane;
    float vx = evec[e * 3 + 0], vy = evec[e * 3 + 1], vz = evec[e * 3 + 2];
    float d = elen[e];
    float invr = frcp(sqrtf(vx * vx + vy * vy + vz * vz) + 1e-8f);
    float x = vx * invr, y = vy * invr, z = vz * invr;
    float* sh = &s_sh[wave][lane * 10];
    sh[0] = 1.f;  sh[1] = y;      sh[2] = z;     sh[3] = x;
    sh[4] = 3.f * z * z - 1.f;    sh[5] = x * z; sh[6] = y * z;
    sh[7] = x * y;                sh[8] = x * x - y * y;
    const float W0 = 3.14159265358979323846f / 6.0f;  // pi/CUTOFF
    float cut = 0.5f * (__cosf(d * W0) + 1.f) * (d < 6.0f ? 1.f : 0.f);
    float invd = frcp(d);
    _Float16* arow = abuf + lane * 32;
#pragma unroll
    for (int i = 0; i < 8; ++i)
      arow[i] = (_Float16)(__sinf(d * (float)(i + 1) * W0) * invd * cut);
#pragma unroll
    for (int i = 8; i < 32; ++i) arow[i] = (_Float16)0.f;
    s_dst[wave][lane] = eidx[N_EDGES + e];
  }
  wait_ds();

  // ---- rad layer 1: 2 x (16x32) @ (32x64), B shared ----
  const v16h* pW1 = (const v16h*)panW1;
  v16h aR0 = lds_loadA(abuf, 32, lane, 0);
  v16h aR1 = lds_loadA(abuf + 16 * 32, 32, lane, 0);
#pragma unroll
  for (int nt = 0; nt < 4; ++nt) {
    v16h b = pW1[nt * 32 + lane];
    v8f c0 = wmma16(aR0, b, zero8());
    v8f c1 = wmma16(aR1, b, zero8());
    float bias = rad_b1[nt * 16 + nL];
#pragma unroll
    for (int j = 0; j < 8; ++j) {
      int m = j + 8 * kh;
      hbuf[m * 64 + nt * 16 + nL] = (_Float16)siluf(c0[j] + bias);
      hbuf[(16 + m) * 64 + nt * 16 + nL] = (_Float16)siluf(c1[j] + bias);
    }
  }
  wait_ds();

  // ---- rad layer 2: 2 x (16x64) @ (64x128), B shared ----
  const v16h* pW2 = (const v16h*)panW2;
  v16h aH[2][2];
#pragma unroll
  for (int t = 0; t < 2; ++t)
#pragma unroll
    for (int kt = 0; kt < 2; ++kt)
      aH[t][kt] = lds_loadA(hbuf + t * 16 * 64, 64, lane, kt * 32);
#pragma unroll
  for (int nt = 0; nt < 8; ++nt) {
    v8f c0 = zero8(), c1 = zero8();
#pragma unroll
    for (int kt = 0; kt < 2; ++kt) {
      v16h b = pW2[(kt * 8 + nt) * 32 + lane];
      c0 = wmma16(aH[0][kt], b, c0);
      c1 = wmma16(aH[1][kt], b, c1);
    }
    float bias = rad_b2[nt * 16 + nL];
#pragma unroll
    for (int j = 0; j < 8; ++j) {
      int m = j + 8 * kh;
      radb[m * 128 + nt * 16 + nL] = (_Float16)siluf(c0[j] + bias);
      radb[(16 + m) * 128 + nt * 16 + nL] = (_Float16)siluf(c1[j] + bias);
    }
  }
  wait_ds();

  // ---- tensor product: edge_msg = sum_s (rad * sh_s) @ W_s, B shared ----
  const v16h* pTP = (const v16h*)panTP;
  v8f acc[2][8];
#pragma unroll
  for (int t = 0; t < 2; ++t)
#pragma unroll
    for (int i = 0; i < 8; ++i) acc[t][i] = zero8();
  for (int s = 0; s < 9; ++s) {
    _Float16 sh0 = (_Float16)s_sh[wave][nL * 10 + s];
    _Float16 sh1 = (_Float16)s_sh[wave][(16 + nL) * 10 + s];
#pragma unroll
    for (int kt = 0; kt < 4; ++kt) {
      const _Float16* p0 = radb + nL * 128 + kt * 32 + kh * 8;
      const _Float16* p1 = radb + (16 + nL) * 128 + kt * 32 + kh * 8;
      v8h lo0 = *(const v8h*)(p0), hi0 = *(const v8h*)(p0 + 16);
      v8h lo1 = *(const v8h*)(p1), hi1 = *(const v8h*)(p1 + 16);
      v16h a0, a1;
#pragma unroll
      for (int i = 0; i < 8; ++i) {
        a0[i] = lo0[i] * sh0; a0[i + 8] = hi0[i] * sh0;
        a1[i] = lo1[i] * sh1; a1[i + 8] = hi1[i] * sh1;
      }
#pragma unroll
      for (int nt = 0; nt < 8; ++nt) {
        v16h b = pTP[(s * 32 + kt * 8 + nt) * 32 + lane];
        acc[0][nt] = wmma16(a0, b, acc[0][nt]);
        acc[1][nt] = wmma16(a1, b, acc[1][nt]);
      }
    }
  }

  // ---- bias + atomic scatter to agg[dst] ----
#pragma unroll
  for (int nt = 0; nt < 8; ++nt) {
    float bias = tp_b[nt * 16 + nL];
#pragma unroll
    for (int t = 0; t < 2; ++t)
#pragma unroll
      for (int j = 0; j < 8; ++j) {
        int m = t * 16 + j + 8 * kh;
        atomicAdd(&agg[(size_t)s_dst[wave][m] * 128 + nt * 16 + nL],
                  acc[t][nt][j] + bias);
      }
  }
}

// ---------------- node update MLP ----------------
__global__ __launch_bounds__(128) void k_node(
    const int* __restrict__ anum, const float* __restrict__ embed,
    const float* __restrict__ agg, const float* __restrict__ msg_b1,
    const float* __restrict__ msg_b2, const _Float16* __restrict__ panW1,
    const _Float16* __restrict__ panW2, float* __restrict__ upd,
    _Float16* __restrict__ upd16) {
  __shared__ _Float16 s_c[4][16 * 192];
  __shared__ _Float16 s_h[4][16 * 128];
  int wave = threadIdx.x >> 5, lane = threadIdx.x & 31;
  int nL = lane & 15, kh = (lane >> 4) & 1;
  int a0 = (blockIdx.x * 4 + wave) * 16;
  _Float16* cbuf = s_c[wave];
  _Float16* hbuf = s_h[wave];

  if (lane < 16) {
    int a = a0 + lane;
    int z = anum[a];
    _Float16* row = cbuf + lane * 192;
    for (int d = 0; d < 64; ++d)  row[d] = (_Float16)embed[z * 64 + d];
    for (int d = 0; d < 128; ++d) row[64 + d] = (_Float16)agg[(size_t)a * 128 + d];
  }
  wait_ds();

  const v16h* pW1 = (const v16h*)panW1;
  v16h aT[6];
#pragma unroll
  for (int kt = 0; kt < 6; ++kt) aT[kt] = lds_loadA(cbuf, 192, lane, kt * 32);
#pragma unroll
  for (int nt = 0; nt < 8; ++nt) {
    v8f c = zero8();
#pragma unroll
    for (int kt = 0; kt < 6; ++kt)
      c = wmma16(aT[kt], pW1[(kt * 8 + nt) * 32 + lane], c);
    float bias = msg_b1[nt * 16 + nL];
#pragma unroll
    for (int j = 0; j < 8; ++j)
      hbuf[(j + 8 * kh) * 128 + nt * 16 + nL] = (_Float16)siluf(c[j] + bias);
  }
  wait_ds();

  const v16h* pW2 = (const v16h*)panW2;
  v16h aU[4];
#pragma unroll
  for (int kt = 0; kt < 4; ++kt) aU[kt] = lds_loadA(hbuf, 128, lane, kt * 32);
#pragma unroll
  for (int nt = 0; nt < 8; ++nt) {
    v8f c = zero8();
#pragma unroll
    for (int kt = 0; kt < 4; ++kt)
      c = wmma16(aU[kt], pW2[(kt * 8 + nt) * 32 + lane], c);
    float bias = msg_b2[nt * 16 + nL];
#pragma unroll
    for (int j = 0; j < 8; ++j) {
      int m = j + 8 * kh;
      float x = c[j] + bias;
      upd[(size_t)(a0 + m) * 128 + nt * 16 + nL] = x;
      upd16[(size_t)(a0 + m) * 128 + nt * 16 + nL] = (_Float16)x;
    }
  }
}

// ---------------- qkv projection ----------------
__global__ __launch_bounds__(128) void k_qkv(
    const _Float16* __restrict__ upd16, const float* __restrict__ attn_b_in,
    const _Float16* __restrict__ panIn, _Float16* __restrict__ qkv16) {
  __shared__ _Float16 s_u[4][16 * 128];
  int wave = threadIdx.x >> 5, lane = threadIdx.x & 31;
  int nL = lane & 15, kh = (lane >> 4) & 1;
  int a0 = (blockIdx.x * 4 + wave) * 16;
  _Float16* ub = s_u[wave];

  if (lane < 16) {
    const uint4* g = (const uint4*)(upd16 + (size_t)(a0 + lane) * 128);
    uint4* d = (uint4*)(ub + lane * 128);
#pragma unroll
    for (int i = 0; i < 16; ++i) d[i] = g[i];
  }
  wait_ds();

  v16h aT[4];
#pragma unroll
  for (int kt = 0; kt < 4; ++kt) aT[kt] = lds_loadA(ub, 128, lane, kt * 32);
  const v16h* pIn = (const v16h*)panIn;
  for (int nt = 0; nt < 24; ++nt) {
    v8f c = zero8();
#pragma unroll
    for (int kt = 0; kt < 4; ++kt)
      c = wmma16(aT[kt], pIn[(kt * 24 + nt) * 32 + lane], c);
    float bias = attn_b_in[nt * 16 + nL];
#pragma unroll
    for (int j = 0; j < 8; ++j) {
      int m = j + 8 * kh;
      qkv16[(size_t)(a0 + m) * 384 + nt * 16 + nL] = (_Float16)(c[j] + bias);
    }
  }
}

// -------- flash attention: 32 queries / wave (2 M-tiles share K/V) --------
__global__ __launch_bounds__(128) void k_attn(const _Float16* __restrict__ qkv16,
                                              _Float16* __restrict__ att16) {
  __shared__ _Float16 s_q[4][32 * 32];
  __shared__ _Float16 s_k[4][32 * 32];
  __shared__ _Float16 s_v[4][32 * 32];   // transposed: [d][key]
  __shared__ _Float16 s_p[4][32 * 32];
  int wave = threadIdx.x >> 5, lane = threadIdx.x & 31;
  int nL = lane & 15, kh = (lane >> 4) & 1;
  int tile = blockIdx.x * 4 + wave;
  int head = tile & 3, q0 = (tile >> 2) * 32;
  _Float16* qb = s_q[wave];
  _Float16* kb = s_k[wave];
  _Float16* vb = s_v[wave];
  _Float16* pb = s_p[wave];

  {
    const uint4* g = (const uint4*)(qkv16 + (size_t)(q0 + lane) * 384 + head * 32);
    uint4* d = (uint4*)(qb + lane * 32);
    d[0] = g[0]; d[1] = g[1];
  }
  wait_ds();
  v16h aQ[2];
#pragma unroll
  for (int t = 0; t < 2; ++t) aQ[t] = lds_loadA(qb + t * 16 * 32, 32, lane, 0);

  float mi[2][8], li[2][8];
  v8f acc[2][2];
#pragma unroll
  for (int t = 0; t < 2; ++t) {
#pragma unroll
    for (int j = 0; j < 8; ++j) { mi[t][j] = -1e30f; li[t][j] = 0.f; }
    acc[t][0] = zero8(); acc[t][1] = zero8();
  }
  const float sc = 0.17677669529663687f;  // 1/sqrt(32)

  for (int kbk = 0; kbk < N_ATOMS / 32; ++kbk) {
    int kk0 = kbk * 32;
    {
      int r = lane;
      const uint4* gk = (const uint4*)(qkv16 + (size_t)(kk0 + r) * 384 + 128 + head * 32);
      uint4* dk = (uint4*)(kb + r * 32);
      dk[0] = gk[0]; dk[1] = gk[1]; dk[2] = gk[2]; dk[3] = gk[3];
      const _Float16* gv = qkv16 + (size_t)(kk0 + r) * 384 + 256 + head * 32;
#pragma unroll
      for (int dd = 0; dd < 32; ++dd) vb[dd * 32 + r] = gv[dd];
    }
    wait_ds();

    v16h bK0 = lds_loadA(kb, 32, lane, 0);
    v16h bK1 = lds_loadA(kb + 16 * 32, 32, lane, 0);
#pragma unroll
    for (int t = 0; t < 2; ++t) {
      v8f s0 = wmma16(aQ[t], bK0, zero8());
      v8f s1 = wmma16(aQ[t], bK1, zero8());
#pragma unroll
      for (int j = 0; j < 8; ++j) {
        float x0 = s0[j] * sc, x1 = s1[j] * sc;
        float v = fmaxf(x0, x1);
#pragma unroll
        for (int off = 1; off < 16; off <<= 1)
          v = fmaxf(v, __shfl_xor(v, off, 32));
        float nm = fmaxf(mi[t][j], v);
        float al = __expf(mi[t][j] - nm);
        float p0 = __expf(x0 - nm), p1 = __expf(x1 - nm);
        float rs = p0 + p1;
#pragma unroll
        for (int off = 1; off < 16; off <<= 1) rs += __shfl_xor(rs, off, 32);
        li[t][j] = li[t][j] * al + rs;
        mi[t][j] = nm;
        acc[t][0][j] *= al; acc[t][1][j] *= al;
        int m = t * 16 + j + 8 * kh;
        pb[m * 32 + nL] = (_Float16)p0;
        pb[m * 32 + 16 + nL] = (_Float16)p1;
      }
    }
    wait_ds();

    v16h bV0 = lds_loadA(vb, 32, lane, 0);
    v16h bV1 = lds_loadA(vb + 16 * 32, 32, lane, 0);
#pragma unroll
    for (int t = 0; t < 2; ++t) {
      v16h aP = lds_loadA(pb + t * 16 * 32, 32, lane, 0);
      acc[t][0] = wmma16(aP, bV0, acc[t][0]);
      acc[t][1] = wmma16(aP, bV1, acc[t][1]);
    }
  }

#pragma unroll
  for (int t = 0; t < 2; ++t)
#pragma unroll
    for (int j = 0; j < 8; ++j) {
      float inv = frcp(li[t][j]);
      int m = t * 16 + j + 8 * kh;
      att16[(size_t)(q0 + m) * 128 + head * 32 + nL] = (_Float16)(acc[t][0][j] * inv);
      att16[(size_t)(q0 + m) * 128 + head * 32 + 16 + nL] = (_Float16)(acc[t][1][j] * inv);
    }
}

// ---------------- output projection / gate / final GEMM ----------------
__global__ __launch_bounds__(128) void k_final(
    const _Float16* __restrict__ att16, const _Float16* __restrict__ upd16,
    const float* __restrict__ attn_b_out, const float* __restrict__ gate_b,
    const float* __restrict__ out_b, const _Float16* __restrict__ panAO,
    const _Float16* __restrict__ panG, const _Float16* __restrict__ panO,
    float* __restrict__ out) {
  __shared__ _Float16 s_a[4][16 * 128];  // att tile, then reused for mix
  __shared__ _Float16 s_u[4][16 * 128];
  int wave = threadIdx.x >> 5, lane = threadIdx.x & 31;
  int nL = lane & 15, kh = (lane >> 4) & 1;
  int a0 = (blockIdx.x * 4 + wave) * 16;
  _Float16* ab = s_a[wave];
  _Float16* ub = s_u[wave];

  if (lane < 16) {
    const uint4* ga = (const uint4*)(att16 + (size_t)(a0 + lane) * 128);
    const uint4* gu = (const uint4*)(upd16 + (size_t)(a0 + lane) * 128);
    uint4* da = (uint4*)(ab + lane * 128);
    uint4* du = (uint4*)(ub + lane * 128);
#pragma unroll
    for (int i = 0; i < 16; ++i) { da[i] = ga[i]; du[i] = gu[i]; }
  }
  wait_ds();

  v16h aA[4], aU[4];
#pragma unroll
  for (int kt = 0; kt < 4; ++kt) {
    aA[kt] = lds_loadA(ab, 128, lane, kt * 32);
    aU[kt] = lds_loadA(ub, 128, lane, kt * 32);
  }

  const v16h* pAO = (const v16h*)panAO;
  const v16h* pG  = (const v16h*)panG;
#pragma unroll
  for (int nt = 0; nt < 8; ++nt) {
    v8f cA = zero8(), cG = zero8();
#pragma unroll
    for (int kt = 0; kt < 4; ++kt) {
      cA = wmma16(aA[kt], pAO[(kt * 8 + nt) * 32 + lane], cA);
      cG = wmma16(aU[kt], pG[(kt * 8 + nt) * 32 + lane], cG);
    }
    float bA = attn_b_out[nt * 16 + nL];
    float bG = gate_b[nt * 16 + nL];
#pragma unroll
    for (int j = 0; j < 8; ++j) {
      int m = j + 8 * kh;
      float att_o = cA[j] + bA;
      float g = sigmf(cG[j] + bG);
      float uval = (float)ub[m * 128 + nt * 16 + nL];
      float mix = g * att_o + (1.f - g) * uval;
      ab[m * 128 + nt * 16 + nL] = (_Float16)mix;  // reuse att buffer
    }
  }
  wait_ds();

  v16h aM[4];
#pragma unroll
  for (int kt = 0; kt < 4; ++kt) aM[kt] = lds_loadA(ab, 128, lane, kt * 32);
  const v16h* pO = (const v16h*)panO;
#pragma unroll
  for (int nt = 0; nt < 8; ++nt) {
    v8f c = zero8();
#pragma unroll
    for (int kt = 0; kt < 4; ++kt)
      c = wmma16(aM[kt], pO[(kt * 8 + nt) * 32 + lane], c);
    float bias = out_b[nt * 16 + nL];
#pragma unroll
    for (int j = 0; j < 8; ++j) {
      int m = j + 8 * kh;
      out[(size_t)(a0 + m) * 128 + nt * 16 + nL] = c[j] + bias;
    }
  }
}

// ---------------- host launcher ----------------
extern "C" void kernel_launch(void* const* d_in, const int* in_sizes, int n_in,
                              void* d_out, int out_size, void* d_ws,
                              size_t ws_size, hipStream_t stream) {
  (void)in_sizes; (void)n_in; (void)out_size; (void)ws_size;
  const int*   anum     = (const int*)d_in[0];
  const int*   eidx     = (const int*)d_in[2];
  const float* evec     = (const float*)d_in[3];
  const float* elen     = (const float*)d_in[4];
  const float* embed    = (const float*)d_in[5];
  const float* rad_w1   = (const float*)d_in[6];
  const float* rad_b1   = (const float*)d_in[7];
  const float* rad_w2   = (const float*)d_in[8];
  const float* rad_b2   = (const float*)d_in[9];
  const float* tp_w     = (const float*)d_in[10];
  const float* tp_b     = (const float*)d_in[11];
  const float* msg_w1   = (const float*)d_in[12];
  const float* msg_b1   = (const float*)d_in[13];
  const float* msg_w2   = (const float*)d_in[14];
  const float* msg_b2   = (const float*)d_in[15];
  const float* attn_w_in  = (const float*)d_in[16];
  const float* attn_b_in  = (const float*)d_in[17];
  const float* attn_w_out = (const float*)d_in[18];
  const float* attn_b_out = (const float*)d_in[19];
  const float* gate_w   = (const float*)d_in[20];
  const float* gate_b   = (const float*)d_in[21];
  const float* out_w    = (const float*)d_in[22];
  const float* out_b    = (const float*)d_in[23];
  float* out = (float*)d_out;

  char* w = (char*)d_ws;
  _Float16* panRadW1 = (_Float16*)(w + OFF_PAN_RADW1);
  _Float16* panRadW2 = (_Float16*)(w + OFF_PAN_RADW2);
  _Float16* panTP    = (_Float16*)(w + OFF_PAN_TP);
  _Float16* panMsgW1 = (_Float16*)(w + OFF_PAN_MSGW1);
  _Float16* panMsgW2 = (_Float16*)(w + OFF_PAN_MSGW2);
  _Float16* panIn    = (_Float16*)(w + OFF_PAN_ATTNIN);
  _Float16* panAO    = (_Float16*)(w + OFF_PAN_ATTNOUT);
  _Float16* panG     = (_Float16*)(w + OFF_PAN_GATE);
  _Float16* panO     = (_Float16*)(w + OFF_PAN_OUT);
  float*    agg      = (float*)(w + OFF_AGG);
  float*    upd      = (float*)(w + OFF_UPD);
  _Float16* upd16    = (_Float16*)(w + OFF_UPD16);
  _Float16* qkv16    = (_Float16*)(w + OFF_QKV16);
  _Float16* att16    = (_Float16*)(w + OFF_ATT16);

  auto packGrid = [](int Ktot, int N) {
    int total = (Ktot / 32) * (N / 16) * 512;
    return dim3((total + 255) / 256);
  };

  // weight packing (cheap; L2-resident afterwards)
  k_pack<<<packGrid(32, 64), 256, 0, stream>>>(panRadW1, rad_w1, 32, 8, 64, 0, 0, 0);
  k_pack<<<packGrid(64, 128), 256, 0, stream>>>(panRadW2, rad_w2, 64, 64, 128, 0, 0, 0);
  for (int s = 0; s < 9; ++s)
    k_pack<<<packGrid(128, 128), 256, 0, stream>>>(panTP + s * 16384, tp_w, 128,
                                                   128, 128, 2, 0, s);
  k_pack<<<packGrid(192, 128), 256, 0, stream>>>(panMsgW1, msg_w1, 192, 192, 128, 0, 0, 0);
  k_pack<<<packGrid(128, 128), 256, 0, stream>>>(panMsgW2, msg_w2, 128, 128, 128, 0, 0, 0);
  k_pack<<<packGrid(128, 384), 256, 0, stream>>>(panIn, attn_w_in, 128, 128, 384, 1, 128, 0);
  k_pack<<<packGrid(128, 128), 256, 0, stream>>>(panAO, attn_w_out, 128, 128, 128, 1, 128, 0);
  k_pack<<<packGrid(128, 128), 256, 0, stream>>>(panG, gate_w, 128, 128, 128, 0, 0, 0);
  k_pack<<<packGrid(128, 128), 256, 0, stream>>>(panO, out_w, 128, 128, 128, 0, 0, 0);

  k_zero<<<(N_ATOMS * HID + 255) / 256, 256, 0, stream>>>(agg, N_ATOMS * HID);

  k_edge<<<N_EDGES / 128, 128, 0, stream>>>(evec, elen, eidx, rad_b1, rad_b2,
                                            tp_b, panRadW1, panRadW2, panTP, agg);
  k_node<<<N_ATOMS / 64, 128, 0, stream>>>(anum, embed, agg, msg_b1, msg_b2,
                                           panMsgW1, panMsgW2, upd, upd16);
  k_qkv<<<N_ATOMS / 64, 128, 0, stream>>>(upd16, attn_b_in, panIn, qkv16);
  k_attn<<<(N_ATOMS / 32), 128, 0, stream>>>(qkv16, att16);
  k_final<<<N_ATOMS / 64, 128, 0, stream>>>(att16, upd16, attn_b_out, gate_b,
                                            out_b, panAO, panG, panO, out);
}
